// Conv2D_37761352466685
// MI455X (gfx1250) — 8-bit matrix operand order repaired and checked by reading, compiled, not run on hardware
//
#include <hip/hip_runtime.h>
#include <stdint.h>

typedef __attribute__((ext_vector_type(8))) int v8i;

#define IN_H 512
#define IN_W 512
#define CIN 32
#define COUT 64
#define OUT_H 510
#define OUT_W 510
#define ROW_BYTES (IN_W * CIN)
#define X_BYTES (IN_H * IN_W * CIN)
#define KREAL 288
#define KPAD 320
#define WFRAG_DWORDS (5 * 4 * 32 * 8)
#define MANT_MAX_I32 2147418112

__global__ __launch_bounds__(256)
void pack_x_kernel(const int* __restrict__ x32, uint32_t* __restrict__ x8) {
    const int i = blockIdx.x * blockDim.x + threadIdx.x;
    const int4 v = ((const int4*)x32)[i];
    x8[i] = (uint32_t)(v.x & 0xFF)        | ((uint32_t)(v.y & 0xFF) << 8) |
            ((uint32_t)(v.z & 0xFF) << 16) | ((uint32_t)(v.w & 0xFF) << 24);
}

__global__ __launch_bounds__(256)
void pack_w_kernel(const int* __restrict__ filt, uint32_t* __restrict__ wf) {
    const int idx = blockIdx.x * blockDim.x + threadIdx.x;
    if (idx >= WFRAG_DWORDS) return;
    const int v    = idx & 7;
    const int lane = (idx >> 3) & 31;
    const int ct   = idx >> 8;
    const int nt   = ct & 3;
    const int kc   = ct >> 2;
    const int n    = nt * 16 + (lane & 15);
    const int half = lane >> 4;
    const int kbase = ((v & 4) ? 32 : 0) + half * 16 + (v & 3) * 4;
    uint32_t d = 0;
    #pragma unroll
    for (int j = 0; j < 4; ++j) {
        const int gk = kc * 64 + kbase + j;
        uint32_t byte = 0u;
        if (gk < KREAL) {
            const int fh = gk / 96;
            const int rem = gk - fh * 96;
            const int fw = rem >> 5;
            const int c  = rem & 31;
            byte = (uint32_t)(filt[((n * 3 + fh) * 3 + fw) * CIN + c] & 0xFF);
        }
        d |= byte << (8 * j);
    }
    wf[idx] = d;
}

__global__ __launch_bounds__(256)
void conv_wmma_kernel(const uint8_t* __restrict__ xp,
                      const int* __restrict__ wf,
                      const int* __restrict__ bias,
                      const int* __restrict__ qmant,
                      const int* __restrict__ expo,
                      const int* __restrict__ zp_ptr,
                      float* __restrict__ out) {
    const int lane   = threadIdx.x & 31;
    const int wave   = threadIdx.x >> 5;
    const int out_y  = blockIdx.y;
    const int tile_x = blockIdx.x * 128;
    const int half   = lane >> 4;

    int px = tile_x + wave * 16 + (lane & 15);
    if (px > OUT_W - 1) px = OUT_W - 1;
    const uint8_t* xrow = xp + (size_t)px * CIN;

    v8i acc[4] = {};

    #pragma unroll
    for (int kc = 0; kc < 5; ++kc) {
        v8i a;
        #pragma unroll
        for (int j = 0; j < 4; ++j) {
            const int gk = kc * 64 + (j >> 1) * 32 + half * 16 + (j & 1) * 8;
            const int fh = (gk >= 192) ? 2 : ((gk >= 96) ? 1 : 0);
            const int r  = gk - fh * 96;
            const uint64_t u =
                *(const uint64_t*)(xrow + (size_t)(out_y + fh) * ROW_BYTES + r);
            a[2 * j]     = (int)(uint32_t)u;
            a[2 * j + 1] = (int)(uint32_t)(u >> 32);
        }
        const v8i* wb = (const v8i*)wf + (size_t)kc * 4 * 32 + lane;
        #pragma unroll
        for (int nt = 0; nt < 4; ++nt) {
            const v8i b = wb[nt * 32];
            acc[nt] = __builtin_amdgcn_wmma_i32_16x16x64_iu8(
                 true, a,  true, b, acc[nt],
                 false,  false);
        }
    }

    const int zp     = zp_ptr[0];
    const int m_base = tile_x + wave * 16 + half * 8;
    #pragma unroll
    for (int nt = 0; nt < 4; ++nt) {
        const int co  = nt * 16 + (lane & 15);
        const int b   = bias[co];
        const int qm  = qmant[co];
        const int red = (qm < MANT_MAX_I32) ? ((qm + (1 << 15)) >> 16) : 32767;
        const int64_t ts  = (int64_t)(15 - expo[co]);
        const int64_t rnd = (int64_t)1 << (ts - 1);
        #pragma unroll
        for (int r = 0; r < 8; ++r) {
            const int mx = m_base + r;
            if (mx < OUT_W) {
                int64_t v = (int64_t)(acc[nt][r] + b) * (int64_t)red + rnd;
                v >>= ts;
                v += (int64_t)zp;
                v = (v < -128) ? -128 : ((v > 127) ? 127 : v);
                out[((size_t)out_y * OUT_W + mx) * COUT + co] = (float)(int)v;
            }
        }
    }
}

extern "C" void kernel_launch(void* const* d_in, const int* in_sizes, int n_in,
                              void* d_out, int out_size, void* d_ws, size_t ws_size,
                              hipStream_t stream) {
    const int* x    = (const int*)d_in[0];
    const int* filt = (const int*)d_in[1];
    const int* bias = (const int*)d_in[2];
    const int* qm   = (const int*)d_in[3];
    const int* ex   = (const int*)d_in[4];
    const int* zp   = (const int*)d_in[5];

    uint8_t*  xp = (uint8_t*)d_ws;
    uint32_t* wf = (uint32_t*)((char*)d_ws + X_BYTES);

    pack_x_kernel<<<(X_BYTES / 4) / 256, 256, 0, stream>>>(x, (uint32_t*)xp);
    pack_w_kernel<<<(WFRAG_DWORDS + 255) / 256, 256, 0, stream>>>(filt, wf);
    conv_wmma_kernel<<<dim3((OUT_W + 127) / 128, OUT_H), 256, 0, stream>>>(
        xp, (const int*)wf, bias, qm, ex, zp, (float*)d_out);
}
